// BoundaryAttentionModule_45114336477799
// MI455X (gfx1250) — matrix operand order repaired and checked by reading, not compiled, not run on hardware
//
#include <hip/hip_runtime.h>
#include <hip/hip_bf16.h>
#include <math.h>

// ---------------------------------------------------------------------------
// BoundaryAttentionModule on MI455X (gfx1250), wave32 + v_wmma_f32_16x16x32_f16
// + Tensor Data Mover (tensor_load_to_lds) streaming of the attention matrix.
//
// Shapes: b=4, c=256, h=w=64 -> HW=4096, mid=64
// Pipeline (all f16 GEMM operands, fp32 accumulation via WMMA):
//   prep:   Ut[p][c]  = f16(U[c][p]);  Kt[p][c] = key-path(boundary);  W->f16
//   proj:   Qt[p][o]  = sum_c Ut[p][c]*Wq[o][c]   (WMMA)
//           Vt[p][o]  = sum_c Ut[p][c]*Wv[o][c]   (WMMA)
//   pass1a: colmax[i] = max_q E[q,i],  E[q,i]=sum_c Qt[q][c]*Kt[i][c]  (WMMA)
//   pass1b: Ap[q][i]  = exp(E[q,i]-colmax[i]) (f16), denom[i]=sum_q    (WMMA)
//   vsc:    Vsc[c][i] = Vt[i][c] / denom[i]
//   pass2:  out[c][j] = gamma * sum_i Vsc[c][i]*Ap[j][i] + U[c][j]
//           (WMMA; B-operand strips TDM-loaded to LDS, double-buffered)
// ---------------------------------------------------------------------------

typedef __attribute__((ext_vector_type(16))) _Float16 v16h;
typedef __attribute__((ext_vector_type(8)))  _Float16 v8h;
typedef __attribute__((ext_vector_type(8)))  float    v8f;
typedef __attribute__((ext_vector_type(4)))  unsigned int v4ui;
typedef __attribute__((ext_vector_type(8)))  int      v8i;
typedef __attribute__((ext_vector_type(4)))  int      v4i;

#define B_   4
#define C_   256
#define HW_  4096
#define MID_ 64

// workspace byte offsets (each buffer 16B+ aligned)
#define OFF_UT   ((size_t)0)                       // f16 [4][4096][256]  8 MiB
#define OFF_KT   ((size_t)(8u  << 20))             // f16 [4][4096][256]  8 MiB
#define OFF_QT   ((size_t)(16u << 20))             // f16 [4][4096][256]  8 MiB
#define OFF_VT   ((size_t)(24u << 20))             // f16 [4][4096][256]  8 MiB
#define OFF_VSC  ((size_t)(32u << 20))             // f16 [4][256][4096]  8 MiB
#define OFF_WQ   ((size_t)(40u << 20))             // f16 [256][256]      128 KiB
#define OFF_WV   (OFF_WQ + 131072)                 // f16 [256][256]      128 KiB
#define OFF_CMAX (OFF_WV + 131072)                 // f32 [4][4096]       64 KiB
#define OFF_DEN  (OFF_CMAX + 65536)                // f32 [4][4096]       64 KiB
#define OFF_AP   (OFF_DEN + 65536)                 // f16 [4][4096][4096] 128 MiB

// -------------------------- WMMA fragment helpers --------------------------

__device__ __forceinline__ v8f zero8() {
  v8f z;
#pragma unroll
  for (int i = 0; i < 8; ++i) z[i] = 0.0f;
  return z;
}

__device__ __forceinline__ v8f wmma_f16(v16h a, v16h b, v8f c) {
  // 8 args: (neg_a, A, neg_b, B, c_mod, C, reuse_a, reuse_b)
  return __builtin_amdgcn_wmma_f32_16x16x32_f16(false, a, false, b, (short)0, c,
                                                false, false);
}

// A-matrix 16x32 f16 fragment from row-major memory tile A[m][k] = base[m*lda+k].
// Lanes 0-15 hold M=lane, k in {0..15}; lanes 16-31 hold M=lane-16,
// k in {16..31}. Two contiguous 16-byte loads per lane.
__device__ __forceinline__ v16h load_a_frag(const _Float16* __restrict__ base,
                                            int lda) {
  int lane  = threadIdx.x & 31;
  int m     = lane & 15;
  int kbase = (lane < 16) ? 0 : 16;
  const _Float16* p = base + (size_t)m * lda + kbase;
  v8h lo = *(const v8h*)(p);
  v8h hi = *(const v8h*)(p + 8);
  v16h a;
#pragma unroll
  for (int i = 0; i < 8; ++i) { a[i] = lo[i]; a[i + 8] = hi[i]; }
  return a;
}

// B-matrix 32x16 f16 fragment where B[k][n] = base[n*ldb + k] (n-major source).
// Lanes 0-15: col n=lane, k=0..15; lanes 16-31: col n=lane-16, k=16..31.
__device__ __forceinline__ v16h load_b_frag(const _Float16* __restrict__ base,
                                            int ldb) {
  int lane = threadIdx.x & 31;
  int n    = lane & 15;
  int klo  = (lane < 16) ? 0 : 16;
  return *(const v16h*)(base + (size_t)n * ldb + klo);
}

// ------------------- Tensor Data Mover: 2-D tile -> LDS ---------------------
// Builds the D# descriptor (groups 0/1 per CDNA5 ISA 8.3/8.4) for a 2-D f16
// tile and issues TENSOR_LOAD_TO_LDS. Groups 2/3 are zero (<=2D tensor).
// Dims/strides in data_size (2-byte) units; addresses in bytes.

__device__ __forceinline__ void tdm_load_2d(unsigned int lds_addr,
                                            unsigned long long gaddr,
                                            unsigned int tensor_d0,
                                            unsigned int tensor_d1,
                                            unsigned int tile_d0,
                                            unsigned int tile_d1,
                                            unsigned long long stride0) {
  v4ui g0;
  g0[0] = 1u;                                              // count=1 (valid)
  g0[1] = lds_addr;                                        // lds_addr[31:0]
  g0[2] = (unsigned int)(gaddr & 0xffffffffull);           // global_addr[31:0]
  g0[3] = (unsigned int)((gaddr >> 32) & 0x01ffffffull) |  // global_addr[56:32]
          (2u << 30);                                      // type=2 ("image")
  v8i g1;
  g1[0] = (int)(1u << 16);                                 // data_size=1 (2B)
  g1[1] = (int)((tensor_d0 & 0xffffu) << 16);              // tensor_dim0[15:0]
  g1[2] = (int)(((tensor_d0 >> 16) & 0xffffu) |
                ((tensor_d1 & 0xffffu) << 16));            // d0[31:16] | d1[15:0]
  g1[3] = (int)(((tensor_d1 >> 16) & 0xffffu) |
                ((tile_d0 & 0xffffu) << 16));              // d1[31:16] | tile_d0
  g1[4] = (int)(tile_d1 & 0xffffu);                        // tile_d1 | tile_d2=0
  g1[5] = (int)(unsigned int)(stride0 & 0xffffffffull);    // dim0_stride[31:0]
  g1[6] = (int)(unsigned int)((stride0 >> 32) & 0xffffull);// [47:32] | d1str=0
  g1[7] = 0;
  v4i z4;
#pragma unroll
  for (int i = 0; i < 4; ++i) z4[i] = 0;
#if __clang_major__ >= 23
  v8i z8;
#pragma unroll
  for (int i = 0; i < 8; ++i) z8[i] = 0;
  __builtin_amdgcn_tensor_load_to_lds(g0, g1, z4, z4, z8, 0);
#else
  __builtin_amdgcn_tensor_load_to_lds(g0, g1, z4, z4, 0);
#endif
}

// ------------------------------- prep kernels -------------------------------

__global__ void prep_ut_k(const float* __restrict__ U, _Float16* __restrict__ Ut) {
  int p = blockIdx.x, b = blockIdx.y, c = threadIdx.x;
  Ut[((size_t)(b * HW_ + p)) * C_ + c] =
      (_Float16)U[((size_t)(b * C_ + c)) * HW_ + p];
}

__global__ void prep_w_k(const float* __restrict__ Wq, const float* __restrict__ Wv,
                         _Float16* __restrict__ Wq16, _Float16* __restrict__ Wv16) {
  int i = blockIdx.x * blockDim.x + threadIdx.x;
  Wq16[i] = (_Float16)Wq[i];
  Wv16[i] = (_Float16)Wv[i];
}

// Key path: nearest-downsampled boundary -> 1x1(1->64) -> BN -> ReLU -> 1x1(64->256)
__global__ void prep_kt_k(const float* __restrict__ boundary,
                          const float* __restrict__ w1, const float* __restrict__ bns,
                          const float* __restrict__ bnb, const float* __restrict__ bnm,
                          const float* __restrict__ bnv, const float* __restrict__ w2,
                          _Float16* __restrict__ Kt) {
  __shared__ float mid[MID_];
  int p = blockIdx.x, b = blockIdx.y, t = threadIdx.x;
  int y = p >> 6, x = p & 63;                       // nearest: src = 2*dst
  float bm = boundary[(size_t)b * 16384 + (size_t)(2 * y) * 128 + 2 * x];
  if (t < MID_) {
    float inv = bns[t] * rsqrtf(bnv[t] + 1e-5f);
    float v   = w1[t] * bm * inv + (bnb[t] - bnm[t] * inv);
    mid[t]    = v > 0.0f ? v : 0.0f;
  }
  __syncthreads();
  float acc = 0.0f;
#pragma unroll
  for (int m = 0; m < MID_; ++m) acc += w2[t * MID_ + m] * mid[m];
  Kt[((size_t)(b * HW_ + p)) * C_ + t] = (_Float16)acc;
}

// ---------------------- projection GEMM: Out[p][o] -------------------------

__global__ void proj_k(const _Float16* __restrict__ Ut,
                       const _Float16* __restrict__ W16,
                       _Float16* __restrict__ Out) {
  int b = blockIdx.y;
  int w = threadIdx.x >> 5, lane = threadIdx.x & 31;
  int half = lane >> 4, n = lane & 15;
  int pbase = blockIdx.x * 64 + (w & 3) * 16;
  int obase = (w >> 2) * 128;
  const _Float16* Ub = Ut + (size_t)b * HW_ * C_;

  v8f acc[8];
#pragma unroll
  for (int j = 0; j < 8; ++j) acc[j] = zero8();

#pragma unroll
  for (int c0 = 0; c0 < C_; c0 += 32) {
    v16h a = load_a_frag(Ub + (size_t)pbase * C_ + c0, C_);
#pragma unroll
    for (int j = 0; j < 8; ++j) {
      v16h bf = load_b_frag(W16 + (size_t)(obase + j * 16) * C_ + c0, C_);
      acc[j] = wmma_f16(a, bf, acc[j]);
    }
  }

  _Float16* Ob = Out + (size_t)b * HW_ * C_;
#pragma unroll
  for (int j = 0; j < 8; ++j)
#pragma unroll
    for (int r = 0; r < 8; ++r)
      Ob[(size_t)(pbase + r + 8 * half) * C_ + obase + j * 16 + n] =
          (_Float16)acc[j][r];
}

// ------------- pass 1a: column max of energy over the q axis ---------------

__global__ void colmax_k(const _Float16* __restrict__ Kt,
                         const _Float16* __restrict__ Qt,
                         float* __restrict__ colmax) {
  __shared__ float red[8][32][8];
  int b = blockIdx.y;
  int w = threadIdx.x >> 5, lane = threadIdx.x & 31;
  int ibase = blockIdx.x * 64 + (w & 3) * 16;
  int qoff  = (w >> 2) * 16;
  const _Float16* Kb = Kt + (size_t)b * HW_ * C_;
  const _Float16* Qb = Qt + (size_t)b * HW_ * C_;

  v16h af[8];                     // stationary K fragments
#pragma unroll
  for (int s = 0; s < 8; ++s)
    af[s] = load_a_frag(Kb + (size_t)ibase * C_ + s * 32, C_);

  float mv[8];
#pragma unroll
  for (int r = 0; r < 8; ++r) mv[r] = -3.0e38f;

  for (int q = qoff; q < HW_; q += 32) {
    v8f acc = zero8();
#pragma unroll
    for (int s = 0; s < 8; ++s) {
      v16h bf = load_b_frag(Qb + (size_t)q * C_ + s * 32, C_);
      acc = wmma_f16(af[s], bf, acc);
    }
#pragma unroll
    for (int r = 0; r < 8; ++r) mv[r] = fmaxf(mv[r], acc[r]);
  }

#pragma unroll
  for (int r = 0; r < 8; ++r) red[w][lane][r] = mv[r];
  __syncthreads();
  if (threadIdx.x < 64) {
    int t = threadIdx.x;
    int isub = t >> 4, half = (t >> 3) & 1, r = t & 7;
    float m = -3.0e38f;
    for (int l = 0; l < 16; ++l) {
      m = fmaxf(m, red[isub][half * 16 + l][r]);
      m = fmaxf(m, red[isub + 4][half * 16 + l][r]);
    }
    colmax[(size_t)b * HW_ + blockIdx.x * 64 + t] = m;
  }
}

// ------ pass 1b: Ap[q][i] = exp(E[q,i]-colmax[i]) (f16) and denom[i] -------

__global__ void expstore_k(const _Float16* __restrict__ Kt,
                           const _Float16* __restrict__ Qt,
                           const float* __restrict__ colmax,
                           _Float16* __restrict__ Ap,
                           float* __restrict__ denom) {
  __shared__ float red[8][32][8];
  int b = blockIdx.y;
  int w = threadIdx.x >> 5, lane = threadIdx.x & 31;
  int half = lane >> 4, n = lane & 15;
  int ibase = blockIdx.x * 64 + (w & 3) * 16;
  int qoff  = (w >> 2) * 16;
  const _Float16* Kb = Kt + (size_t)b * HW_ * C_;
  const _Float16* Qb = Qt + (size_t)b * HW_ * C_;
  _Float16* Ab = Ap + (size_t)b * HW_ * HW_;

  v16h af[8];
#pragma unroll
  for (int s = 0; s < 8; ++s)
    af[s] = load_a_frag(Kb + (size_t)ibase * C_ + s * 32, C_);

  float mvv[8], dsum[8];
#pragma unroll
  for (int r = 0; r < 8; ++r) {
    mvv[r]  = colmax[(size_t)b * HW_ + ibase + 8 * half + r];
    dsum[r] = 0.0f;
  }

  for (int q = qoff; q < HW_; q += 32) {
    v8f acc = zero8();
#pragma unroll
    for (int s = 0; s < 8; ++s) {
      v16h bf = load_b_frag(Qb + (size_t)q * C_ + s * 32, C_);
      acc = wmma_f16(af[s], bf, acc);
    }
    v8h ph;
#pragma unroll
    for (int r = 0; r < 8; ++r) {
      float p = __expf(acc[r] - mvv[r]);
      dsum[r] += p;
      ph[r] = (_Float16)p;
    }
    // row-major A'[q][i]; 8 contiguous f16 per lane (i = ibase + 8*half + r)
    *(v8h*)(Ab + (size_t)(q + n) * HW_ + ibase + 8 * half) = ph;
  }

#pragma unroll
  for (int r = 0; r < 8; ++r) red[w][lane][r] = dsum[r];
  __syncthreads();
  if (threadIdx.x < 64) {
    int t = threadIdx.x;
    int isub = t >> 4, h2 = (t >> 3) & 1, r = t & 7;
    float s = 0.0f;
    for (int l = 0; l < 16; ++l) {
      s += red[isub][h2 * 16 + l][r];
      s += red[isub + 4][h2 * 16 + l][r];
    }
    denom[(size_t)b * HW_ + blockIdx.x * 64 + t] = s;
  }
}

// -------------------- Vsc[c][i] = Vt[i][c] / denom[i] ----------------------

__global__ void vsc_k(const _Float16* __restrict__ Vt, const float* __restrict__ denom,
                      _Float16* __restrict__ Vsc) {
  int i = blockIdx.x, b = blockIdx.y, c = threadIdx.x;
  float d = denom[(size_t)b * HW_ + i];
  float v = (float)Vt[((size_t)(b * HW_ + i)) * C_ + c];
  Vsc[((size_t)(b * C_ + c)) * HW_ + i] = (_Float16)(v / d);
}

// --------- pass 2: out[c][j] = gamma * sum_i Vsc[c][i]*Ap[j][i] + U --------
// GEMM with M=c, N=j, K=i (4096). The 64x32 f16 B-strip (attention rows) per
// K-step is identical for all 8 waves: wave 0 TDM-loads it into a double-
// buffered LDS tile (tensor_load_to_lds + s_wait_tensorcnt); everyone reads
// B fragments from LDS.

__global__ void out_k(const _Float16* __restrict__ Vsc,
                      const _Float16* __restrict__ Ap,
                      const float* __restrict__ U,
                      const float* __restrict__ gamma,
                      float* __restrict__ Out) {
  __shared__ alignas(64) _Float16 abuf[2][64 * 32];   // 2 x 4KB B-strips
  int b = blockIdx.y;
  int w = threadIdx.x >> 5, lane = threadIdx.x & 31;
  int half = lane >> 4, n = lane & 15;
  int jmac  = blockIdx.x * 64;
  int jbase = jmac + (w & 3) * 16;
  int cbase = (w >> 2) * 128;
  int klo   = (lane < 16) ? 0 : 16;
  const _Float16* Vb = Vsc + (size_t)b * C_ * HW_;
  const _Float16* Ab = Ap + (size_t)b * HW_ * HW_;

  v8f acc[8];
#pragma unroll
  for (int mt = 0; mt < 8; ++mt) acc[mt] = zero8();

  const unsigned int lds0 = (unsigned int)(uintptr_t)(&abuf[0][0]);
  const unsigned int lds1 = (unsigned int)(uintptr_t)(&abuf[1][0]);
  const unsigned long long gbase =
      (unsigned long long)(uintptr_t)(Ab + (size_t)jmac * HW_);
  const int NSTEP = HW_ / 32;   // 128

  if (w == 0)   // prime buffer 0 with the i0=0 strip
    tdm_load_2d(lds0, gbase, HW_, HW_, 32, 64, HW_);

  for (int t = 0; t < NSTEP; ++t) {
    int i0 = t * 32;
    if (w == 0) {
      if (t + 1 < NSTEP) {
        tdm_load_2d(((t + 1) & 1) ? lds1 : lds0,
                    gbase + (unsigned long long)(i0 + 32) * 2,
                    HW_, HW_, 32, 64, HW_);
        __builtin_amdgcn_s_wait_tensorcnt(1);  // current strip landed
      } else {
        __builtin_amdgcn_s_wait_tensorcnt(0);
      }
    }
    __syncthreads();                           // strip visible to all waves

    v16h bf = *(const v16h*)(&abuf[t & 1][((w & 3) * 16 + n) * 32 + klo]);
#pragma unroll
    for (int mt = 0; mt < 8; ++mt) {
      v16h a = load_a_frag(Vb + (size_t)(cbase + mt * 16) * HW_ + i0, HW_);
      acc[mt] = wmma_f16(a, bf, acc[mt]);
    }
    __syncthreads();                           // done reading before overwrite
  }

  float g = gamma[0];
  const float* Ub = U + (size_t)b * C_ * HW_;
  float* Ob = Out + (size_t)b * C_ * HW_;
  int j = jbase + n;
#pragma unroll
  for (int mt = 0; mt < 8; ++mt)
#pragma unroll
    for (int r = 0; r < 8; ++r) {
      int c = cbase + mt * 16 + r + 8 * half;
      size_t idx = (size_t)c * HW_ + j;
      Ob[idx] = g * acc[mt][r] + Ub[idx];
    }
}

// ------------------------------- launch -------------------------------------

extern "C" void kernel_launch(void* const* d_in, const int* in_sizes, int n_in,
                              void* d_out, int out_size, void* d_ws, size_t ws_size,
                              hipStream_t stream) {
  (void)in_sizes; (void)n_in; (void)out_size; (void)ws_size;
  const float* boundary = (const float*)d_in[0];
  const float* U        = (const float*)d_in[1];
  const float* key_w1   = (const float*)d_in[2];
  const float* bns      = (const float*)d_in[3];
  const float* bnb      = (const float*)d_in[4];
  const float* bnm      = (const float*)d_in[5];
  const float* bnv      = (const float*)d_in[6];
  const float* key_w2   = (const float*)d_in[7];
  const float* Wq       = (const float*)d_in[8];
  const float* Wv       = (const float*)d_in[9];
  const float* gamma    = (const float*)d_in[10];
  float* out = (float*)d_out;

  char* ws = (char*)d_ws;
  _Float16* Ut   = (_Float16*)(ws + OFF_UT);
  _Float16* Kt   = (_Float16*)(ws + OFF_KT);
  _Float16* Qt   = (_Float16*)(ws + OFF_QT);
  _Float16* Vt   = (_Float16*)(ws + OFF_VT);
  _Float16* Vsc  = (_Float16*)(ws + OFF_VSC);
  _Float16* Wq16 = (_Float16*)(ws + OFF_WQ);
  _Float16* Wv16 = (_Float16*)(ws + OFF_WV);
  float*    cmax = (float*)(ws + OFF_CMAX);
  float*    den  = (float*)(ws + OFF_DEN);
  _Float16* Ap   = (_Float16*)(ws + OFF_AP);

  dim3 blk(256);
  prep_ut_k  <<<dim3(HW_, B_), blk, 0, stream>>>(U, Ut);
  prep_w_k   <<<dim3(256),     blk, 0, stream>>>(Wq, Wv, Wq16, Wv16);
  prep_kt_k  <<<dim3(HW_, B_), blk, 0, stream>>>(boundary, key_w1, bns, bnb, bnm,
                                                 bnv, key_w2, Kt);
  proj_k     <<<dim3(64, B_),  blk, 0, stream>>>(Ut, Wq16, Qt);
  proj_k     <<<dim3(64, B_),  blk, 0, stream>>>(Ut, Wv16, Vt);
  colmax_k   <<<dim3(64, B_),  blk, 0, stream>>>(Kt, Qt, cmax);
  expstore_k <<<dim3(64, B_),  blk, 0, stream>>>(Kt, Qt, cmax, Ap, den);
  vsc_k      <<<dim3(HW_, B_), blk, 0, stream>>>(Vt, den, Vsc);
  out_k      <<<dim3(64, B_),  blk, 0, stream>>>(Vsc, Ap, U, gamma, out);
}
